// MHSA_23467701305478
// MI455X (gfx1250) — matrix operand order repaired and checked by reading, not compiled, not run on hardware
//
#include <hip/hip_runtime.h>
#include <hip/hip_bf16.h>

// ---------------------------------------------------------------------------
// MHSA on MI455X (gfx1250): bf16 WMMA everywhere, fp32 accumulate + softmax.
// B=8, N=1024, C=1024, H=16, hd=64.
// GEMMs use CDNA5 async global->LDS copies with double-buffered LDS tiles.
// ---------------------------------------------------------------------------

typedef __bf16 bf16_t;
typedef __attribute__((ext_vector_type(16))) __bf16 bf16x16;
typedef __attribute__((ext_vector_type(8)))  float  v8f;

union Frag16 {
    bf16x16      v;
    unsigned int u[8];
    bf16_t       h[16];
};

__device__ __forceinline__ v8f wmma_bf16(bf16x16 a, bf16x16 b, v8f c) {
    return __builtin_amdgcn_wmma_f32_16x16x32_bf16(
        /*neg_a=*/false, a, /*neg_b=*/false, b,
        /*c_mod=*/(short)0, c, /*reuse_a=*/false, /*reuse_b=*/false);
}

// Async 16B global->LDS copy. INST_OFFSET is applied to both the LDS and the
// global address (ISA 10.x async pseudocode), so one base covers 4 slices.
template <int OFF>
__device__ __forceinline__ void async_ld_b128(unsigned lds_off, const void* g) {
    asm volatile("global_load_async_to_lds_b128 %0, %1, off offset:%2"
                 :: "v"(lds_off), "v"(g), "n"(OFF)
                 : "memory");
}

template <int N>
__device__ __forceinline__ void wait_async() {
    asm volatile("s_wait_asynccnt %0" :: "n"(N) : "memory");
}

// low 32 bits of a generic pointer to __shared__ == LDS byte offset
__device__ __forceinline__ unsigned lds_addr(const void* p) {
    return (unsigned)(size_t)p;
}

// A-fragment (16x32, M x K): lane = M (lane%16); per-lane 8 dword reads.
// K(pair p) = 2p + (p>=4 ? 8 : 0) + (lane>=16 ? 8 : 0)   [ISA 7.12.2]
template <int STRIDE>
__device__ __forceinline__ bf16x16 load_frag_a(const bf16_t* base, int row,
                                               int kbase, int lane) {
    Frag16 f;
    const int laneoff = (lane >= 16) ? 8 : 0;
    const bf16_t* rp = base + (size_t)row * STRIDE + kbase;
#pragma unroll
    for (int p = 0; p < 8; ++p) {
        int kk = 2 * p + ((p >= 4) ? 8 : 0) + laneoff;
        f.u[p] = *(const unsigned int*)(rp + kk);
    }
    return f.v;
}

// B-fragment (32x16, K x N) read from a [N][K]-major tile (i.e. B^T row-major):
// lane = N (lane%16); K(pair p) = kbase + 2p + (p>=4 ? 8 : 0) + (lane>=16 ? 8 : 0).
template <int STRIDE>
__device__ __forceinline__ bf16x16 load_frag_b(const bf16_t* base, int ncol,
                                               int kbase, int lane) {
    Frag16 f;
    const int k0 = kbase + ((lane >= 16) ? 8 : 0);
    const bf16_t* rp = base + (size_t)ncol * STRIDE + k0;
#pragma unroll
    for (int p = 0; p < 8; ++p) {
        f.u[p] = *(const unsigned int*)(rp + 2 * p + ((p >= 4) ? 8 : 0));
    }
    return f.v;
}

// ---------------------------------------------------------------------------
// Stage 1: fp32 -> bf16 conversion (4-wide)
// ---------------------------------------------------------------------------
__global__ void cvt_f32_to_bf16_x4(const float4* __restrict__ src,
                                   bf16_t* __restrict__ dst, int n4) {
    int i = blockIdx.x * blockDim.x + threadIdx.x;
    if (i < n4) {
        float4 f = src[i];
        union { bf16_t h[4]; uint2 u; } r;
        r.h[0] = (bf16_t)f.x; r.h[1] = (bf16_t)f.y;
        r.h[2] = (bf16_t)f.z; r.h[3] = (bf16_t)f.w;
        *(uint2*)(dst + (size_t)i * 4) = r.u;
    }
}

// ---------------------------------------------------------------------------
// GEMM tiles: block 128x128x64, 256 threads (8 waves), wave tile 32x64.
// Double-buffered LDS, async global->LDS staging.
// ---------------------------------------------------------------------------
#define BM 128
#define BN 128
#define BK2 64

// ---------------------------------------------------------------------------
// Stage 2: QKV projection. y[m, d] = sum_c x[m, c] * Wqkv[d, c]
// M = 8192, Nout = 3072, K = 1024. Epilogue scatters into Q/K/V [B,H,N,hd].
// ---------------------------------------------------------------------------
__global__ __launch_bounds__(256) void qkv_gemm_kernel(
    const bf16_t* __restrict__ xb,    // [8192, 1024]
    const bf16_t* __restrict__ wb,    // [3072, 1024]
    bf16_t* __restrict__ qout,        // [8,16,1024,64]
    bf16_t* __restrict__ kout,
    bf16_t* __restrict__ vout) {
    __shared__ bf16_t As[2][BM][BK2];
    __shared__ bf16_t Bs[2][BN][BK2];

    const int tid  = threadIdx.x;
    const int lane = tid & 31;
    const int wave = tid >> 5;   // 0..7
    const int wr   = wave & 3;   // 4 row-waves * 32 rows
    const int wc   = wave >> 2;  // 2 col-waves * 64 cols
    const int m0   = blockIdx.y * BM;
    const int n0   = blockIdx.x * BN;
    const int l16  = lane & 15;

    v8f acc[2][4] = {};

    const int r    = tid >> 1;        // 0..127
    const int sseg = (tid & 1) * 32;  // 0 or 32 (elements)

    const bf16_t* ga0 = xb + (size_t)(m0 + r) * 1024 + sseg;
    const bf16_t* gb0 = wb + (size_t)(n0 + r) * 1024 + sseg;

    auto stage = [&](int buf, int kk) {
        const bf16_t* ga = ga0 + kk;
        const bf16_t* gb = gb0 + kk;
        unsigned la = lds_addr(&As[buf][r][sseg]);
        unsigned lb = lds_addr(&Bs[buf][r][sseg]);
        async_ld_b128<0>(la, ga);   async_ld_b128<16>(la, ga);
        async_ld_b128<32>(la, ga);  async_ld_b128<48>(la, ga);
        async_ld_b128<0>(lb, gb);   async_ld_b128<16>(lb, gb);
        async_ld_b128<32>(lb, gb);  async_ld_b128<48>(lb, gb);
    };

    stage(0, 0);
    for (int i = 0; i < 16; ++i) {
        const int cur = i & 1;
        if (i < 15) {
            stage(1 - cur, (i + 1) * BK2);
            wait_async<8>();   // retire only the current buffer's 8 requests
        } else {
            wait_async<0>();
        }
        __syncthreads();

#pragma unroll
        for (int kc = 0; kc < 2; ++kc) {
            bf16x16 af[2], bfrag[4];
#pragma unroll
            for (int ii = 0; ii < 2; ++ii)
                af[ii] = load_frag_a<BK2>(&As[cur][0][0],
                                          wr * 32 + ii * 16 + l16, kc * 32, lane);
#pragma unroll
            for (int j = 0; j < 4; ++j)
                bfrag[j] = load_frag_b<BK2>(&Bs[cur][0][0],
                                            wc * 64 + j * 16 + l16, kc * 32, lane);
#pragma unroll
            for (int ii = 0; ii < 2; ++ii)
#pragma unroll
                for (int j = 0; j < 4; ++j)
                    acc[ii][j] = wmma_bf16(af[ii], bfrag[j], acc[ii][j]);
        }
        __syncthreads();  // reads done before next prefetch overwrites buffer
    }

    // Epilogue: C-tile layout (VGPR g -> row g (+8 for lanes>=16), col = lane%16)
    const int hl = (lane >= 16) ? 8 : 0;
#pragma unroll
    for (int i = 0; i < 2; ++i) {
#pragma unroll
        for (int j = 0; j < 4; ++j) {
            const int n   = n0 + wc * 64 + j * 16 + l16;   // [0,3072)
            const int t   = n >> 10;                        // 0:q 1:k 2:v
            const int rem = n & 1023;
            const int h   = rem >> 6;
            const int e   = rem & 63;
            bf16_t* dst = (t == 0) ? qout : (t == 1) ? kout : vout;
#pragma unroll
            for (int g = 0; g < 8; ++g) {
                const int m  = m0 + wr * 32 + i * 16 + hl + g;  // [0,8192)
                const int b  = m >> 10;
                const int nn = m & 1023;
                dst[(((size_t)(b * 16 + h) * 1024) + nn) * 64 + e] =
                    (bf16_t)acc[i][j][g];
            }
        }
    }
}

// ---------------------------------------------------------------------------
// Stage 3: flash attention. grid = (B*H, N/128), 256 threads (8 waves),
// each wave owns 16 query rows; streams 64-key K/V blocks via LDS.
// K block staged with async global->LDS; V transposed through VGPRs.
// ---------------------------------------------------------------------------
__global__ __launch_bounds__(256) void attn_kernel(
    const bf16_t* __restrict__ q,   // [B*H, 1024, 64]
    const bf16_t* __restrict__ k,
    const bf16_t* __restrict__ v,
    bf16_t* __restrict__ attnout) { // [B, 1024, 1024] bf16
    __shared__ bf16_t Ks[64][64];        // [key][hd]
    __shared__ bf16_t Vt[64][64];        // [hd][key]  (transposed at stage)
    __shared__ bf16_t Ps[8][16][64];     // per-wave P tile [qrow][key]

    const int bh   = blockIdx.x;         // 0..127
    const int qblk = blockIdx.y;         // 0..7
    const int b    = bh >> 4;
    const int h    = bh & 15;
    const int tid  = threadIdx.x;
    const int lane = tid & 31;
    const int wave = tid >> 5;
    const int l16  = lane & 15;
    const int hl   = (lane >= 16) ? 8 : 0;

    const size_t base = (size_t)bh * 1024 * 64;

    // Q fragments stay in registers for the whole kernel (16x64 -> 2 A-frags)
    bf16x16 qf[2];
    {
        const int qrow = qblk * 128 + wave * 16 + l16;
#pragma unroll
        for (int f = 0; f < 2; ++f)
            qf[f] = load_frag_a<64>(q + base, qrow, f * 32, lane);
    }

    float mrow[8], lrow[8];
    v8f o[4] = {};
#pragma unroll
    for (int g = 0; g < 8; ++g) { mrow[g] = -1e30f; lrow[g] = 0.0f; }

    const int sr = tid >> 2;            // 0..63 staging row
    const int ss = (tid & 3) * 16;      // staging col segment

    for (int kb = 0; kb < 16; ++kb) {
        __syncthreads();  // previous iteration's LDS reads done
        {
            const size_t g0 = base + (size_t)(kb * 64 + sr) * 64 + ss;
            // K: async straight into LDS (row-major, B-fragment friendly)
            const unsigned lk = lds_addr(&Ks[sr][ss]);
            async_ld_b128<0>(lk, k + g0);
            async_ld_b128<16>(lk, k + g0);
            // V: through VGPRs with transpose into Vt[hd][key]
            Frag16 vv;
            const uint4* pv = (const uint4*)(v + g0);
            *(uint4*)&vv.u[0] = pv[0]; *(uint4*)&vv.u[4] = pv[1];
#pragma unroll
            for (int i = 0; i < 16; ++i) Vt[ss + i][sr] = vv.h[i];
        }
        wait_async<0>();
        __syncthreads();

        // S = (Q K^T) * scale, 4 tiles of 16x16
        v8f sT[4];
#pragma unroll
        for (int j = 0; j < 4; ++j) {
            v8f c = {};
#pragma unroll
            for (int ks = 0; ks < 2; ++ks) {
                bf16x16 kf = load_frag_b<64>(&Ks[0][0], j * 16 + l16, ks * 32, lane);
                c = wmma_bf16(qf[ks], kf, c);
            }
            sT[j] = c * 0.125f;  // hd^-0.5
        }

        // online softmax (row stats per C-layout slot g; 16-lane reductions)
        float alpha[8];
#pragma unroll
        for (int g = 0; g < 8; ++g) {
            float mx = fmaxf(fmaxf(sT[0][g], sT[1][g]), fmaxf(sT[2][g], sT[3][g]));
#pragma unroll
            for (int off = 8; off >= 1; off >>= 1)
                mx = fmaxf(mx, __shfl_xor(mx, off, 32));
            float mn = fmaxf(mrow[g], mx);
            alpha[g] = __expf(mrow[g] - mn);
            mrow[g]  = mn;
            float rs = 0.0f;
#pragma unroll
            for (int j = 0; j < 4; ++j) {
                float p = __expf(sT[j][g] - mn);
                sT[j][g] = p;
                rs += p;
            }
#pragma unroll
            for (int off = 8; off >= 1; off >>= 1)
                rs += __shfl_xor(rs, off, 32);
            lrow[g] = lrow[g] * alpha[g] + rs;
        }
#pragma unroll
        for (int jo = 0; jo < 4; ++jo)
#pragma unroll
            for (int g = 0; g < 8; ++g) o[jo][g] *= alpha[g];

        // C-layout -> A-layout relayout of P through per-wave LDS
#pragma unroll
        for (int j = 0; j < 4; ++j)
#pragma unroll
            for (int g = 0; g < 8; ++g)
                Ps[wave][g + hl][j * 16 + l16] = (bf16_t)sT[j][g];

        // intra-wave LDS store->load ordering (per-wave region, no barrier)
        asm volatile("s_wait_dscnt 0" ::: "memory");

        // O += P V
#pragma unroll
        for (int ks = 0; ks < 2; ++ks) {
            bf16x16 pf = load_frag_a<64>(&Ps[wave][0][0], l16, ks * 32, lane);
#pragma unroll
            for (int jo = 0; jo < 4; ++jo) {
                bf16x16 vf = load_frag_b<64>(&Vt[0][0], jo * 16 + l16, ks * 32, lane);
                o[jo] = wmma_bf16(pf, vf, o[jo]);
            }
        }
    }

    // normalize and store to [B, N, C] bf16
#pragma unroll
    for (int jo = 0; jo < 4; ++jo) {
#pragma unroll
        for (int g = 0; g < 8; ++g) {
            const int row = qblk * 128 + wave * 16 + hl + g;
            const int col = h * 64 + jo * 16 + l16;
            attnout[((size_t)b * 1024 + row) * 1024 + col] =
                (bf16_t)(o[jo][g] / lrow[g]);
        }
    }
}

// ---------------------------------------------------------------------------
// Stage 4: output projection + bias. out[m, n] = sum_c a[m,c]*Wp[n,c] + bias[n]
// M = 8192, Nout = 1024, K = 1024. fp32 epilogue to d_out.
// ---------------------------------------------------------------------------
__global__ __launch_bounds__(256) void proj_gemm_kernel(
    const bf16_t* __restrict__ ab,    // [8192, 1024]
    const bf16_t* __restrict__ wb,    // [1024, 1024]
    const float* __restrict__ bias,   // [1024]
    float* __restrict__ out) {        // [8192, 1024]
    __shared__ bf16_t As[2][BM][BK2];
    __shared__ bf16_t Bs[2][BN][BK2];

    const int tid  = threadIdx.x;
    const int lane = tid & 31;
    const int wave = tid >> 5;
    const int wr   = wave & 3;
    const int wc   = wave >> 2;
    const int m0   = blockIdx.y * BM;
    const int n0   = blockIdx.x * BN;
    const int l16  = lane & 15;

    v8f acc[2][4] = {};

    const int r    = tid >> 1;
    const int sseg = (tid & 1) * 32;

    const bf16_t* ga0 = ab + (size_t)(m0 + r) * 1024 + sseg;
    const bf16_t* gb0 = wb + (size_t)(n0 + r) * 1024 + sseg;

    auto stage = [&](int buf, int kk) {
        const bf16_t* ga = ga0 + kk;
        const bf16_t* gb = gb0 + kk;
        unsigned la = lds_addr(&As[buf][r][sseg]);
        unsigned lb = lds_addr(&Bs[buf][r][sseg]);
        async_ld_b128<0>(la, ga);   async_ld_b128<16>(la, ga);
        async_ld_b128<32>(la, ga);  async_ld_b128<48>(la, ga);
        async_ld_b128<0>(lb, gb);   async_ld_b128<16>(lb, gb);
        async_ld_b128<32>(lb, gb);  async_ld_b128<48>(lb, gb);
    };

    stage(0, 0);
    for (int i = 0; i < 16; ++i) {
        const int cur = i & 1;
        if (i < 15) {
            stage(1 - cur, (i + 1) * BK2);
            wait_async<8>();
        } else {
            wait_async<0>();
        }
        __syncthreads();

#pragma unroll
        for (int kc = 0; kc < 2; ++kc) {
            bf16x16 af[2], bfrag[4];
#pragma unroll
            for (int ii = 0; ii < 2; ++ii)
                af[ii] = load_frag_a<BK2>(&As[cur][0][0],
                                          wr * 32 + ii * 16 + l16, kc * 32, lane);
#pragma unroll
            for (int j = 0; j < 4; ++j)
                bfrag[j] = load_frag_b<BK2>(&Bs[cur][0][0],
                                            wc * 64 + j * 16 + l16, kc * 32, lane);
#pragma unroll
            for (int ii = 0; ii < 2; ++ii)
#pragma unroll
                for (int j = 0; j < 4; ++j)
                    acc[ii][j] = wmma_bf16(af[ii], bfrag[j], acc[ii][j]);
        }
        __syncthreads();
    }

    const int hl = (lane >= 16) ? 8 : 0;
#pragma unroll
    for (int j = 0; j < 4; ++j) {
        const int n = n0 + wc * 64 + j * 16 + l16;
        const float bv = bias[n];
#pragma unroll
        for (int i = 0; i < 2; ++i) {
#pragma unroll
            for (int g = 0; g < 8; ++g) {
                const int m = m0 + wr * 32 + i * 16 + hl + g;
                out[(size_t)m * 1024 + n] = acc[i][j][g] + bv;
            }
        }
    }
}

// ---------------------------------------------------------------------------
// Launch
// ---------------------------------------------------------------------------
extern "C" void kernel_launch(void* const* d_in, const int* in_sizes, int n_in,
                              void* d_out, int out_size, void* d_ws, size_t ws_size,
                              hipStream_t stream) {
    const float* x     = (const float*)d_in[0];  // [8,1024,1024]
    const float* Wqkv  = (const float*)d_in[1];  // [3072,1024]
    const float* Wproj = (const float*)d_in[2];  // [1024,1024]
    const float* bproj = (const float*)d_in[3];  // [1024]
    float* out = (float*)d_out;

    char* ws = (char*)d_ws;
    bf16_t* xb     = (bf16_t*)(ws + (size_t)0);          // 16 MB
    bf16_t* wqkvb  = (bf16_t*)(ws + ((size_t)16 << 20)); //  6 MB
    bf16_t* wprojb = (bf16_t*)(ws + ((size_t)22 << 20)); //  2 MB
    bf16_t* qb     = (bf16_t*)(ws + ((size_t)24 << 20)); // 16 MB
    bf16_t* kb     = (bf16_t*)(ws + ((size_t)40 << 20)); // 16 MB
    bf16_t* vb     = (bf16_t*)(ws + ((size_t)56 << 20)); // 16 MB
    bf16_t* attnb  = (bf16_t*)(ws + ((size_t)72 << 20)); // 16 MB (total 88 MB)

    // Stage 1: conversions
    cvt_f32_to_bf16_x4<<<8192, 256, 0, stream>>>((const float4*)x, xb,
                                                 8 * 1024 * 1024 / 4);
    cvt_f32_to_bf16_x4<<<3072, 256, 0, stream>>>((const float4*)Wqkv, wqkvb,
                                                 3 * 1024 * 1024 / 4);
    cvt_f32_to_bf16_x4<<<1024, 256, 0, stream>>>((const float4*)Wproj, wprojb,
                                                 1024 * 1024 / 4);

    // Stage 2: QKV GEMM  (grid: 3072/128 x 8192/128)
    qkv_gemm_kernel<<<dim3(24, 64), 256, 0, stream>>>(xb, wqkvb, qb, kb, vb);

    // Stage 3: attention (grid: B*H x N/128)
    attn_kernel<<<dim3(128, 8), 256, 0, stream>>>(qb, kb, vb, attnb);

    // Stage 4: output projection (grid: 1024/128 x 8192/128)
    proj_gemm_kernel<<<dim3(8, 64), 256, 0, stream>>>(attnb, wprojb, bproj, out);
}
